// SparseFeedForward_22763326668988
// MI455X (gfx1250) — matrix operand order repaired and checked by reading, not compiled, not run on hardware
//
#include <hip/hip_runtime.h>
#include <hip/hip_bf16.h>
#include <math.h>

// Problem constants (B, L, H, Dh, M) = (4, 2048, 16, 256, 1024)
#define BL      8192            // B*L rows
#define HEADS   16
#define DH      256             // per-head dim (K of GEMM1, N of GEMM2)
#define MM      1024            // hidden dim
#define DMODEL  (HEADS * DH)    // 4096
#define ROWS    64              // rows per workgroup
#define MCHUNK  64              // hidden cols per iteration (2x bf16 WMMA K)
#define NCHUNK  (MM / MCHUNK)   // 16
#define B1ELEMS (DH * MCHUNK)   // 16384 bf16 = 32 KB per chunk image
#define B2ELEMS (MCHUNK * DH)   // 16384 bf16 = 32 KB per chunk image

typedef __attribute__((ext_vector_type(16))) __bf16 v16bf;
typedef __attribute__((ext_vector_type(8)))  float  v8f;
typedef unsigned int v4u __attribute__((ext_vector_type(4)));
typedef int          v8i __attribute__((ext_vector_type(8)));
typedef int          v4i __attribute__((ext_vector_type(4)));

#if __has_include(<hip/amd_detail/amd_gfx1250_TDM.h>)
#define TDM_6ARG 1
#endif

// round-to-nearest-even f32 -> bf16
__device__ __forceinline__ __bf16 f2bf(float f) {
  unsigned u = __builtin_bit_cast(unsigned, f);
  u += 0x7FFFu + ((u >> 16) & 1u);
  unsigned short h = (unsigned short)(u >> 16);
  return __builtin_bit_cast(__bf16, h);
}

// 1-D contiguous TDM copy: global -> LDS, nbytes must be a multiple of 8.
// D# per CDNA5 ISA: 2D tile (n8 x 1), data_size=8B.
__device__ __forceinline__ void tdm_load_1d(unsigned lds_addr, const void* gsrc,
                                            unsigned nbytes) {
  unsigned long long ga = (unsigned long long)(uintptr_t)gsrc;
  unsigned n8 = nbytes >> 3;                       // 8-byte units
  v4u g0;
  g0[0] = 1u;                                      // count=1, user descriptor
  g0[1] = lds_addr;                                // lds_addr [63:32]
  g0[2] = (unsigned)(ga & 0xFFFFFFFFu);            // global_addr [95:64]
  g0[3] = (unsigned)((ga >> 32) & 0x1FFFFFFu)      // global_addr [120:96]
        | (2u << 30);                              // type=2 ("image")
  v8i g1;
  g1[0] = (int)(3u << 16);                         // wg_mask=0, data_size=3 (8B)
  g1[1] = (int)((n8 & 0xFFFFu) << 16);             // tensor_dim0[15:0] @ bits[63:48]
  g1[2] = (int)((n8 >> 16) & 0xFFFFu)              // tensor_dim0[31:16]
        | (1 << 16);                               // tensor_dim1 = 1
  g1[3] = (int)((n8 & 0xFFFFu) << 16);             // tile_dim0 = n8 @ bits[127:112]
  g1[4] = 1;                                       // tile_dim1 = 1, tile_dim2 = 0
  g1[5] = (int)n8;                                 // tensor_dim0_stride[31:0]
  g1[6] = 0;                                       // stride hi, dim1_stride lo
  g1[7] = 0;
  v4i z4 = {0, 0, 0, 0};
#ifdef TDM_6ARG
  v8i z8 = {0, 0, 0, 0, 0, 0, 0, 0};
  __builtin_amdgcn_tensor_load_to_lds(g0, g1, z4, z4, z8, 0);
#else
  __builtin_amdgcn_tensor_load_to_lds(g0, g1, z4, z4, 0);
#endif
}

// ---------------- pre-pass: fp32 weights -> bf16 WMMA B-fragment images ----------------
// one block per (head, chunk); image is the exact 32 KB LDS byte layout the GEMMs read.
__global__ __launch_bounds__(256)
void convert_weights_kernel(const float* __restrict__ up,
                            const float* __restrict__ down,
                            __bf16* __restrict__ wB1,
                            __bf16* __restrict__ wB2) {
  const int tid  = threadIdx.x;
  const int blk  = blockIdx.x;                     // head*NCHUNK + mc
  const int head = blk >> 4;
  const int m0   = (blk & (NCHUNK - 1)) * MCHUNK;
  const float* up_h = up   + (size_t)head * DH * MM;
  const float* dn_h = down + (size_t)head * MM * DH;
  __bf16* o1 = wB1 + (size_t)blk * B1ELEMS;
  __bf16* o2 = wB2 + (size_t)blk * B2ELEMS;

  // up chunk (256 x 64) -> B fragments (8 k-steps x 4 col-tiles)
  // B-fragment layout (16-bit B 32x16): lane = (N&15) + 16*((K>>3)&1); elem = (K&7)|((K&16)>>1)
  #pragma unroll 4
  for (int e = tid; e < B1ELEMS; e += 256) {
    int k = e >> 6, j = e & 63;                    // coalesced over j
    float v = up_h[(size_t)k * MM + m0 + j];
    int ks = k >> 5, kl = k & 31;
    int ct = j >> 4, nl = j & 15;
    int ln = nl + (((kl >> 3) & 1) << 4);
    int el = (kl & 7) | ((kl & 16) >> 1);
    o1[(((ks * 4 + ct) * 32 + ln) << 4) + el] = f2bf(v);
  }
  // down chunk (64 x 256) -> B fragments (2 k-steps x 16 col-tiles)
  #pragma unroll 4
  for (int e = tid; e < B2ELEMS; e += 256) {
    int r = e >> 8, d = e & 255;                   // coalesced over d
    float v = dn_h[(size_t)(m0 + r) * DH + d];
    int ks2 = r >> 5, rl2 = r & 31;
    int ct  = d >> 4, nl  = d & 15;
    int ln = nl + (((rl2 >> 3) & 1) << 4);
    int el = (rl2 & 7) | ((rl2 & 16) >> 1);
    o2[(((ks2 * 16 + ct) * 32 + ln) << 4) + el] = f2bf(v);
  }
}

// ---------------- main fused FFN kernel ----------------
__global__ __launch_bounds__(256)
void sparse_ffn_wmma_kernel(const float* __restrict__ x,
                            const __bf16* __restrict__ wB1,
                            const __bf16* __restrict__ wB2,
                            const int*   __restrict__ perm,
                            float* __restrict__ out)
{
  // 160 KB LDS (WGP pool is 320 KB -> 2 workgroups per WGP)
  __shared__ alignas(32) __bf16 sA [4 * 8 * 512];  // 32 KB: A fragments (4 row-tiles x 8 k-steps)
  __shared__ alignas(32) __bf16 sB1[2][B1ELEMS];   // 2 x 32 KB: up-chunk B fragments (TDM dest);
                                                   //   first 8 KB of active buf reused for hidden
  __shared__ alignas(32) __bf16 sB2[2][B2ELEMS];   // 2 x 32 KB: down-chunk B fragments (TDM dest)

  const int tid  = threadIdx.x;
  const int lane = tid & 31;
  const int w    = tid >> 5;                       // wave id 0..7
  const int head = blockIdx.x & (HEADS - 1);
  const int row0 = (blockIdx.x >> 4) * ROWS;

  const int*    perm_h = perm + head * DH;
  const __bf16* w1_h   = wB1 + (size_t)head * NCHUNK * B1ELEMS;
  const __bf16* w2_h   = wB2 + (size_t)head * NCHUNK * B2ELEMS;

  // ---- stage gathered A tile (64 x 256) as bf16 WMMA A-fragments ----
  // A-fragment layout (16-bit A 16x32): lane = (row&15) + 16*((K>>3)&1); elem = (K&7)|((K&16)>>1)
  #pragma unroll 4
  for (int idx = tid; idx < ROWS * DH; idx += 256) {
    int row = idx >> 8;
    int d   = idx & 255;
    float v = x[(size_t)(row0 + row) * DMODEL + perm_h[d]];
    int rt = row >> 4, rl = row & 15;
    int ks = d >> 5,   kk = d & 31;
    int ln = rl + (((kk >> 3) & 1) << 4);
    int el = (kk & 7) | ((kk & 16) >> 1);
    sA[(((rt * 8 + ks) * 32 + ln) << 4) + el] = f2bf(v);
  }

  // ---- prologue: wave 0 launches chunk 0's DMAs into buffer 0 ----
  if (w == 0) {
    tdm_load_1d((unsigned)(uintptr_t)(const void*)&sB1[0][0], w1_h, B1ELEMS * 2);
    tdm_load_1d((unsigned)(uintptr_t)(const void*)&sB2[0][0], w2_h, B2ELEMS * 2);
  }

  // per-wave output accumulators: 2 row-tiles x 4 col-tiles of 16x16 f32
  v8f acc[8];
  #pragma unroll
  for (int i = 0; i < 8; ++i)
    acc[i] = (v8f){0.f, 0.f, 0.f, 0.f, 0.f, 0.f, 0.f, 0.f};

  const int rt1 = w >> 1;          // GEMM1: this wave's hidden row-tile
  const int ct1 = w & 1;           // GEMM1: this wave's hidden col-tile pair
  const int ort = (w >> 2) << 1;   // GEMM2: base row-tile (2 per wave)
  const int oct = (w & 3) << 2;    // GEMM2: base col-tile (4 per wave)

  for (int mc = 0; mc < NCHUNK; ++mc) {
    const int cur = mc & 1;

    // wave 0: kick off next chunk's DMAs (into the other buffer, which the
    // end-of-previous-iteration barrier guaranteed free), then wait until only
    // the 2 just-issued ops remain outstanding (TDM is in-order per wave).
    if (w == 0) {
      if (mc + 1 < NCHUNK) {
        const int nxt = (mc + 1) & 1;
        tdm_load_1d((unsigned)(uintptr_t)(const void*)&sB1[nxt][0],
                    w1_h + (size_t)(mc + 1) * B1ELEMS, B1ELEMS * 2);
        tdm_load_1d((unsigned)(uintptr_t)(const void*)&sB2[nxt][0],
                    w2_h + (size_t)(mc + 1) * B2ELEMS, B2ELEMS * 2);
        __builtin_amdgcn_s_wait_tensorcnt(2);
      } else {
        __builtin_amdgcn_s_wait_tensorcnt(0);
      }
    }
    __syncthreads();   // chunk mc data visible to all waves (also covers sA on mc==0)

    // ---- GEMM1: hidden(64x64) = A(64x256) @ up_chunk(256x64); 2 tiles per wave ----
    v8f hacc0 = (v8f){0.f, 0.f, 0.f, 0.f, 0.f, 0.f, 0.f, 0.f};
    v8f hacc1 = hacc0;
    #pragma unroll
    for (int ks = 0; ks < 8; ++ks) {
      v16bf a  = *(const v16bf*)&sA [(((rt1 * 8 + ks) * 32 + lane) << 4)];
      v16bf b0 = *(const v16bf*)&sB1[cur][(((ks * 4 + ct1 * 2 + 0) * 32 + lane) << 4)];
      v16bf b1 = *(const v16bf*)&sB1[cur][(((ks * 4 + ct1 * 2 + 1) * 32 + lane) << 4)];
      hacc0 = __builtin_amdgcn_wmma_f32_16x16x32_bf16(false, a, false, b0,
                                                      (short)0, hacc0, false, false);
      hacc1 = __builtin_amdgcn_wmma_f32_16x16x32_bf16(false, a, false, b1,
                                                      (short)0, hacc1, false, false);
    }
    __syncthreads();   // all waves done reading sB1[cur] before hidden overwrites its head

    // ---- exact GeLU, restage hidden as bf16 A-fragments into sB1[cur][0..4095] ----
    __bf16* sH = &sB1[cur][0];
    #pragma unroll
    for (int t = 0; t < 2; ++t) {
      const v8f& hv = t ? hacc1 : hacc0;
      int ctl = ct1 * 2 + t;
      #pragma unroll
      for (int v = 0; v < 8; ++v) {
        float hx = hv[v];
        float g  = 0.5f * hx * (1.0f + erff(hx * 0.70710678118654752f));
        int rl  = v + ((lane >> 4) << 3);          // row within 16x16 D tile
        int kk  = ctl * 16 + (lane & 15);          // K2 index within chunk (0..63)
        int ks2 = kk >> 5, kl2 = kk & 31;
        int ln = rl + (((kl2 >> 3) & 1) << 4);
        int el = (kl2 & 7) | ((kl2 & 16) >> 1);
        sH[(((rt1 * 2 + ks2) * 32 + ln) << 4) + el] = f2bf(g);
      }
    }
    __syncthreads();

    // ---- GEMM2: out(64x256) += hidden(64x64) @ down_chunk(64x256) ----
    #pragma unroll
    for (int ks2 = 0; ks2 < 2; ++ks2) {
      #pragma unroll
      for (int i = 0; i < 8; ++i) {
        int rt = ort + (i >> 2);
        int ct = oct + (i & 3);
        v16bf a = *(const v16bf*)&sH [(((rt * 2 + ks2) * 32 + lane) << 4)];
        v16bf b = *(const v16bf*)&sB2[cur][(((ks2 * 16 + ct) * 32 + lane) << 4)];
        acc[i] = __builtin_amdgcn_wmma_f32_16x16x32_bf16(false, a, false, b,
                                                         (short)0, acc[i], false, false);
      }
    }
    __syncthreads();   // buffer `cur` free for the DMA issued at the top of mc+1
  }

  // ---- scatter: out[:, perm[head*256 + d]] = result[:, d] ----
  #pragma unroll
  for (int i = 0; i < 8; ++i) {
    int rt = ort + (i >> 2);
    int ct = oct + (i & 3);
    int d  = ct * 16 + (lane & 15);
    int colg = perm_h[d];
    #pragma unroll
    for (int v = 0; v < 8; ++v) {
      int R = row0 + rt * 16 + v + ((lane >> 4) << 3);
      out[(size_t)R * DMODEL + colg] = acc[i][v];
    }
  }
}

extern "C" void kernel_launch(void* const* d_in, const int* in_sizes, int n_in,
                              void* d_out, int out_size, void* d_ws, size_t ws_size,
                              hipStream_t stream) {
  const float* x    = (const float*)d_in[0];
  const float* up   = (const float*)d_in[1];
  const float* down = (const float*)d_in[2];
  const int*   perm = (const int*)d_in[3];
  float* out = (float*)d_out;

  __bf16* wB1 = (__bf16*)d_ws;                               // 8 MB
  __bf16* wB2 = wB1 + (size_t)HEADS * NCHUNK * B1ELEMS;      // 8 MB

  convert_weights_kernel<<<dim3(HEADS * NCHUNK), dim3(256), 0, stream>>>(up, down, wB1, wB2);
  sparse_ffn_wmma_kernel<<<dim3((BL / ROWS) * HEADS), dim3(256), 0, stream>>>(x, wB1, wB2, perm, out);
}
